// MCA_1305670058667
// MI455X (gfx1250) — matrix operand order repaired and checked by reading, compiled, not run on hardware
//
#include <hip/hip_runtime.h>
#include <hip/hip_bf16.h>
#include <cstdint>
#include <cstddef>

#define B_   16
#define HW_  9216          // 96*96 tokens per batch
#define M_TOK (B_*HW_)     // 147456 rows
#define NCH  96            // scan chunks per sequence
#define CL   96            // chunk length (NCH*CL == HW_)

typedef __attribute__((ext_vector_type(16))) __bf16 v16bf;
typedef __attribute__((ext_vector_type(8)))  float  v8f;

union BfFrag { unsigned short u[16]; uint4 q[2]; v16bf v; };

__device__ __forceinline__ unsigned short f2bf(float f) {
  unsigned int u = __float_as_uint(f);
  unsigned int r = u + 0x7fffu + ((u >> 16) & 1u);   // round-to-nearest-even
  return (unsigned short)(r >> 16);
}
__device__ __forceinline__ float silu_f(float x) {
  // x * rcp(1+exp(-x)) : v_exp_f32 + v_rcp_f32, no IEEE divide sequence
  return x * __builtin_amdgcn_rcpf(1.0f + __expf(-x));
}
__device__ __forceinline__ float softplus_f(float x) {
  return x > 20.f ? x : __logf(1.f + __expf(x));
}

// ---------------- NCHW -> (b, L, 64) bf16, LDS-tiled transpose ----------------
__global__ void k_pre(const float* __restrict__ x, unsigned short* __restrict__ xs_bf) {
  __shared__ float tile[64][33];
  int b  = blockIdx.x / (HW_/32);
  int l0 = (blockIdx.x % (HW_/32)) * 32;
  int tid = threadIdx.x;
  for (int i = 0; i < 8; ++i) {
    int idx = tid + i*256;
    int c = idx >> 5, lo = idx & 31;              // coalesced along lo
    tile[c][lo] = x[((size_t)b*64 + c)*HW_ + l0 + lo];
  }
  __syncthreads();
  for (int i = 0; i < 8; ++i) {
    int idx = tid + i*256;
    int lo = idx >> 6, c = idx & 63;              // coalesced along c
    xs_bf[((size_t)b*HW_ + l0 + lo)*64 + c] = f2bf(tile[c][lo]);
  }
}

// ---------------- WMMA GEMM:  Y = A(bf16, Mx64) @ [W0;W1]^T  ----------------
// W0: N0x64 (optional silu, fp32 and/or bf16 out), W1: N1x64 raw fp32 out.
// N0/N1 in {0,64,128}. Block = 128 threads = 4 waves, 64-row M tile.
// Output staged in LDS then stored fully coalesced as b128/b64.
__global__ void __launch_bounds__(128) k_gemm(
    const unsigned short* __restrict__ A,
    const float* __restrict__ W0, int N0, int act0,
    float* __restrict__ out0, unsigned short* __restrict__ out0b,
    const float* __restrict__ W1, int N1, float* __restrict__ out1)
{
  __shared__ unsigned short ldsW[192*64];     // bf16 weights (<=192 cols)
  __shared__ float ldsO[64*200];              // 64-row tile, 200-f pitch (conflict-free)
  int tid  = threadIdx.x;
  int lane = tid & 31, wid = tid >> 5;
  int half = lane >> 4, l16 = lane & 15;
  size_t mBlock = (size_t)blockIdx.x * 64;

  const unsigned short* ar = A + (mBlock + (size_t)wid*16 + l16) * 64;
  __builtin_prefetch(ar, 0, 1);               // global_prefetch_b8 while we fill LDS

  int Ntot = N0 + N1;
  for (int idx = tid; idx < Ntot*64; idx += 128) {
    int n = idx >> 6, k = idx & 63;
    float f = (n < N0) ? W0[n*64 + k] : W1[(n - N0)*64 + k];
    ldsW[idx] = f2bf(f);
  }
  __syncthreads();

  // A fragments: 16-bit A 16x32 layout (ISA 7.12.2): per lane-half two 16B runs
  BfFrag a0, a1;
  a0.q[0] = *(const uint4*)(ar +      half*8);
  a0.q[1] = *(const uint4*)(ar + 16 + half*8);
  a1.q[0] = *(const uint4*)(ar + 32 + half*8);
  a1.q[1] = *(const uint4*)(ar + 48 + half*8);

  int NT = Ntot >> 4;
  for (int nt = 0; nt < NT; ++nt) {
    // B fragments: 32x16 layout -> lane n, per lane-half two 16B runs (K = half*8 + e, 16 + half*8 + e)
    const unsigned short* br = &ldsW[(nt*16 + l16) * 64 + half*8];
    BfFrag b0, b1;
    b0.q[0] = *(const uint4*)(br);
    b0.q[1] = *(const uint4*)(br + 16);
    b1.q[0] = *(const uint4*)(br + 32);
    b1.q[1] = *(const uint4*)(br + 48);

    v8f acc = {};
    acc = __builtin_amdgcn_wmma_f32_16x16x32_bf16(false, a0.v, false, b0.v,
                                                  (short)0, acc, false, false);
    acc = __builtin_amdgcn_wmma_f32_16x16x32_bf16(false, a1.v, false, b1.v,
                                                  (short)0, acc, false, false);

    int col = nt*16 + l16;
    bool doAct = act0 && (nt*16 < N0);        // uniform per nt (N0 multiple of 16)
    #pragma unroll
    for (int r = 0; r < 8; ++r) {
      float v = acc[r];
      if (doAct) v = silu_f(v);
      ldsO[(wid*16 + half*8 + r)*200 + col] = v;  // D layout: M = r + 8*half
    }
  }
  __syncthreads();

  // ---- coalesced store phase (b128 / b64) ----
  if (out0) {
    int sh = (N0 == 64) ? 6 : 7, msk = N0 - 1;
    for (int i = tid*4; i < 64*N0; i += 512) {
      int row = i >> sh, col = i & msk;
      float4 v = *(const float4*)&ldsO[row*200 + col];
      *(float4*)&out0[(mBlock + row)*(size_t)N0 + col] = v;
    }
  }
  if (out0b) {                                   // only used with N0 == 64
    for (int i = tid*4; i < 64*64; i += 512) {
      int row = i >> 6, col = i & 63;
      const float* s = &ldsO[row*200 + col];
      ushort4 u;
      u.x = f2bf(s[0]); u.y = f2bf(s[1]); u.z = f2bf(s[2]); u.w = f2bf(s[3]);
      *(ushort4*)&out0b[(mBlock + row)*64 + col] = u;
    }
  }
  if (out1) {                                    // N1 == 128
    for (int i = tid*4; i < 64*N1; i += 512) {
      int row = i >> 7, col = i & 127;
      float4 v = *(const float4*)&ldsO[row*200 + N0 + col];
      *(float4*)&out1[(mBlock + row)*(size_t)N1 + col] = v;
    }
  }
}

// ---------------- causal depthwise conv(4) + bias + silu ----------------
__global__ void k_conv(const float* __restrict__ xz, const float* __restrict__ cw,
                       const float* __restrict__ cb, float* __restrict__ xc)
{
  size_t tid = (size_t)blockIdx.x * blockDim.x + threadIdx.x;
  int c = tid & 63;
  size_t t = tid >> 6;
  int l = (int)(t % HW_);
  size_t bbase = t - l;                        // b*HW_
  float s = cb[c];
  if (l >= 3) s += cw[c*4+0] * xz[(bbase + l - 3)*128 + c];
  if (l >= 2) s += cw[c*4+1] * xz[(bbase + l - 2)*128 + c];
  if (l >= 1) s += cw[c*4+2] * xz[(bbase + l - 1)*128 + c];
  s += cw[c*4+3] * xz[(bbase + l)*128 + c];
  xc[tid] = silu_f(s);
}

// ---------------- x_dbl = xc @ xproj^T ; dt = softplus(x_dbl[:,:4]@dtw^T + b) ----
__global__ void k_xdbl(const float* __restrict__ xc, const float* __restrict__ xproj,
                       const float* __restrict__ dtw, const float* __restrict__ dtb,
                       float* __restrict__ x_dbl, float* __restrict__ dt)
{
  __shared__ float sxc[4][64];
  __shared__ float sdbl[4][8];
  int tid = threadIdx.x;
  int tl = tid >> 6, d = tid & 63;
  size_t tok = (size_t)blockIdx.x * 4 + tl;
  sxc[tl][d] = xc[tok*64 + d];
  __syncthreads();
  if (tid < 32) {
    int t2 = tid >> 3, j = tid & 7;
    const float* wr = xproj + j*64;
    float s = 0.f;
    #pragma unroll 8
    for (int k = 0; k < 64; ++k) s += sxc[t2][k] * wr[k];
    sdbl[t2][j] = s;
    x_dbl[((size_t)blockIdx.x*4 + t2)*8 + j] = s;
  }
  __syncthreads();
  float acc = dtb[d];
  #pragma unroll
  for (int r = 0; r < 4; ++r) acc += sdbl[tl][r] * dtw[d*4 + r];
  dt[tok*64 + d] = softplus_f(acc);
}

// ---------------- chunked parallel selective scan ----------------
// phase 1: per (b, chunk, d): affine summary (P = prod a_t, S = folded sum)
__global__ void k_scan1(const float* __restrict__ dt, const float* __restrict__ x_dbl,
                        const float* __restrict__ xc, const float* __restrict__ A_log,
                        float* __restrict__ PS)
{
  int tid = blockIdx.x * blockDim.x + threadIdx.x;
  int d  = tid & 63;
  int ch = (tid >> 6) % NCH;
  int b  = tid / (64 * NCH);
  float A0 = -__expf(A_log[d*2+0]);
  float A1 = -__expf(A_log[d*2+1]);
  float P0=1.f,S0=0.f,P1=1.f,S1=0.f;
  size_t base = (size_t)b * HW_ + (size_t)ch * CL;
  for (int t = 0; t < CL; ++t) {
    size_t p = base + t;
    float dtv = dt[p*64 + d];
    float xcv = xc[p*64 + d];
    const float* xd = x_dbl + p*8;
    float bu = dtv * xcv;
    float a0 = __expf(dtv * A0), a1 = __expf(dtv * A1);
    S0 = fmaf(a0, S0, bu * xd[4]); P0 *= a0;
    S1 = fmaf(a1, S1, bu * xd[5]); P1 *= a1;
  }
  size_t o = (((size_t)b*64 + d)*NCH + ch)*4;
  PS[o]=P0; PS[o+1]=S0; PS[o+2]=P1; PS[o+3]=S1;
}

// phase 2: scan chunk summaries -> per-chunk initial states
__global__ void k_scan2(const float* __restrict__ PS, float* __restrict__ HI)
{
  int tid = blockIdx.x * blockDim.x + threadIdx.x;
  if (tid >= B_*64) return;
  float h0=0.f, h1=0.f;
  size_t base = (size_t)tid * NCH;
  for (int ch = 0; ch < NCH; ++ch) {
    size_t o = base + ch;
    HI[o*2] = h0; HI[o*2+1] = h1;
    h0 = fmaf(PS[o*4+0], h0, PS[o*4+1]);
    h1 = fmaf(PS[o*4+2], h1, PS[o*4+3]);
  }
}

// phase 3: replay chunk with true h_init; fuse C-contraction, D-skip, silu(z) gate;
// emit bf16 rows ready to be the A operand of the out_w WMMA GEMM.
__global__ void k_scan3(const float* __restrict__ dt, const float* __restrict__ x_dbl,
                        const float* __restrict__ xc, const float* __restrict__ xz,
                        const float* __restrict__ A_log, const float* __restrict__ Dp,
                        const float* __restrict__ HI, unsigned short* __restrict__ ybf)
{
  int tid = blockIdx.x * blockDim.x + threadIdx.x;
  int d  = tid & 63;
  int ch = (tid >> 6) % NCH;
  int b  = tid / (64 * NCH);
  float A0 = -__expf(A_log[d*2+0]);
  float A1 = -__expf(A_log[d*2+1]);
  float Dv = Dp[d];
  size_t hi = ((size_t)b*64 + d)*NCH + ch;
  float h0 = HI[hi*2], h1 = HI[hi*2+1];
  size_t base = (size_t)b * HW_ + (size_t)ch * CL;
  for (int t = 0; t < CL; ++t) {
    size_t p = base + t;
    float dtv = dt[p*64 + d];
    float xcv = xc[p*64 + d];
    const float* xd = x_dbl + p*8;
    float bu = dtv * xcv;
    h0 = fmaf(__expf(dtv * A0), h0, bu * xd[4]);
    h1 = fmaf(__expf(dtv * A1), h1, bu * xd[5]);
    float y = h0 * xd[6] + h1 * xd[7] + xcv * Dv;
    float z = xz[p*128 + 64 + d];
    ybf[p*64 + d] = f2bf(y * silu_f(z));
  }
}

// ---------------- layernorm over 64 channels (1 wave32 per token, 2 ch/lane) ----
__global__ void k_ln(const float* __restrict__ in, const float* __restrict__ g,
                     const float* __restrict__ bt, float* __restrict__ out,
                     unsigned short* __restrict__ outb)
{
  int lane = threadIdx.x & 31;
  size_t tok = (size_t)blockIdx.x * 8 + (threadIdx.x >> 5);
  const float* r = in + tok*64;
  float v0 = r[lane], v1 = r[lane + 32];
  float s = v0 + v1;
  for (int off = 16; off > 0; off >>= 1) s += __shfl_xor(s, off, 32);
  float mu = s * (1.f/64.f);
  float d0 = v0 - mu, d1 = v1 - mu;
  float q = d0*d0 + d1*d1;
  for (int off = 16; off > 0; off >>= 1) q += __shfl_xor(q, off, 32);
  float inv = rsqrtf(q * (1.f/64.f) + 1e-5f);
  float o0 = d0*inv*g[lane]    + bt[lane];
  float o1 = d1*inv*g[lane+32] + bt[lane+32];
  out[tok*64+lane]    = o0;
  out[tok*64+lane+32] = o1;
  if (outb) { outb[tok*64+lane] = f2bf(o0); outb[tok*64+lane+32] = f2bf(o1); }
}

// ---------------- silu(res + y) and (b,L,C) -> NCHW, LDS-tiled ----------------
__global__ void k_epi(const float* __restrict__ res, const float* __restrict__ y,
                      float* __restrict__ out)
{
  __shared__ float tile[64][33];
  int b  = blockIdx.x / (HW_/32);
  int l0 = (blockIdx.x % (HW_/32)) * 32;
  int tid = threadIdx.x;
  for (int i = 0; i < 8; ++i) {
    int idx = tid + i*256;
    int lo = idx >> 6, c = idx & 63;
    size_t p = ((size_t)b*HW_ + l0 + lo)*64 + c;
    tile[c][lo] = silu_f(res[p] + y[p]);
  }
  __syncthreads();
  for (int i = 0; i < 8; ++i) {
    int idx = tid + i*256;
    int c = idx >> 5, lo = idx & 31;
    out[((size_t)b*64 + c)*HW_ + l0 + lo] = tile[c][lo];
  }
}

extern "C" void kernel_launch(void* const* d_in, const int* in_sizes, int n_in,
                              void* d_out, int out_size, void* d_ws, size_t ws_size,
                              hipStream_t stream)
{
  (void)in_sizes; (void)n_in; (void)out_size; (void)ws_size;
  const float* x     = (const float*)d_in[0];
  const float* W_res = (const float*)d_in[1];
  const float* m_in_w[2]   = {(const float*)d_in[2],  (const float*)d_in[11]};
  const float* m_conv_w[2] = {(const float*)d_in[3],  (const float*)d_in[12]};
  const float* m_conv_b[2] = {(const float*)d_in[4],  (const float*)d_in[13]};
  const float* m_xproj[2]  = {(const float*)d_in[5],  (const float*)d_in[14]};
  const float* m_dtw[2]    = {(const float*)d_in[6],  (const float*)d_in[15]};
  const float* m_dtb[2]    = {(const float*)d_in[7],  (const float*)d_in[16]};
  const float* m_Alog[2]   = {(const float*)d_in[8],  (const float*)d_in[17]};
  const float* m_D[2]      = {(const float*)d_in[9],  (const float*)d_in[18]};
  const float* m_out_w[2]  = {(const float*)d_in[10], (const float*)d_in[19]};
  const float* ln1_g = (const float*)d_in[20];
  const float* ln1_b = (const float*)d_in[21];
  const float* W_mid = (const float*)d_in[22];
  const float* ln2_g = (const float*)d_in[23];
  const float* ln2_b = (const float*)d_in[24];
  float* out = (float*)d_out;

  char* ws = (char*)d_ws;
  size_t cur = 0;
  auto carve = [&](size_t bytes) -> void* {
    void* p = ws + cur;
    cur += (bytes + 255) & ~(size_t)255;
    return p;
  };
  const size_t M = M_TOK;
  float*          res  = (float*)carve(M*64*4);
  unsigned short* BA   = (unsigned short*)carve(M*64*2);   // bf16 ping
  unsigned short* BB   = (unsigned short*)carve(M*64*2);   // bf16 pong
  float*          xz   = (float*)carve(M*128*4);
  float*          xc   = (float*)carve(M*64*4);
  float*          xdbl = (float*)carve(M*8*4);
  float*          dt   = (float*)carve(M*64*4);
  float*          t0   = (float*)carve(M*64*4);
  float*          t1   = (float*)carve(M*64*4);
  float*          PS   = (float*)carve((size_t)B_*64*NCH*4*4);
  float*          HI   = (float*)carve((size_t)B_*64*NCH*2*4);

  dim3 b256(256), b128(128);
  const int trBlocks = B_ * (HW_/32);
  const int scanT = B_ * 64 * NCH;

  k_pre<<<trBlocks, b256, 0, stream>>>(x, BA);
  // res = silu(xs@W_res^T); xz = xs@in_w1^T  (dual-N WMMA GEMM, N=192)
  k_gemm<<<M/64, b128, 0, stream>>>(BA, W_res, 64, 1, res, nullptr, m_in_w[0], 128, xz);

  auto mamba = [&](int i, unsigned short* ybf) {
    k_conv <<<(M*64)/256, b256, 0, stream>>>(xz, m_conv_w[i], m_conv_b[i], xc);
    k_xdbl <<<M/4,        b256, 0, stream>>>(xc, m_xproj[i], m_dtw[i], m_dtb[i], xdbl, dt);
    k_scan1<<<scanT/256,  b256, 0, stream>>>(dt, xdbl, xc, m_Alog[i], PS);
    k_scan2<<<(B_*64+255)/256, b256, 0, stream>>>(PS, HI);
    k_scan3<<<scanT/256,  b256, 0, stream>>>(dt, xdbl, xc, xz, m_Alog[i], m_D[i], HI, ybf);
  };

  mamba(0, BB);
  k_gemm<<<M/64, b128, 0, stream>>>(BB, m_out_w[0], 64, 0, t0, nullptr, nullptr, 0, nullptr);
  k_ln  <<<M/8,  b256, 0, stream>>>(t0, ln1_g, ln1_b, t1, BA);
  k_gemm<<<M/64, b128, 0, stream>>>(BA, W_mid, 64, 1, nullptr, BB, nullptr, 0, nullptr);
  k_gemm<<<M/64, b128, 0, stream>>>(BB, m_in_w[1], 128, 0, xz, nullptr, nullptr, 0, nullptr);
  mamba(1, BA);
  k_gemm<<<M/64, b128, 0, stream>>>(BA, m_out_w[1], 64, 0, t0, nullptr, nullptr, 0, nullptr);
  k_ln  <<<M/8,  b256, 0, stream>>>(t0, ln2_g, ln2_b, t1, nullptr);
  k_epi <<<trBlocks, b256, 0, stream>>>(res, t1, out);
}
